// SubclassBEVFusionFuserDecoder_30176440222152
// MI455X (gfx1250) — matrix operand order repaired and checked by reading, not compiled, not run on hardware
//
#include <hip/hip_runtime.h>
#include <hip/hip_bf16.h>
#include <math.h>

// ---------------------------------------------------------------------------
// Types for CDNA5 WMMA (wave32): V_WMMA_F32_16X16X32_BF16
//   A: 16x32 bf16 (16 bf16 per lane), B: 32x16 bf16, C/D: 16x16 f32 (8 per lane)
// ---------------------------------------------------------------------------
typedef __attribute__((ext_vector_type(16))) __bf16 bf16x16;
typedef __attribute__((ext_vector_type(8)))  float  f32x8;

#define HW   32400
#define WD   180
#define HD   180
#define HID  128
#define NCLS 10
#define KQ   200
#define KQP  208      /* 200 padded to a 16 multiple for guard-free B loads */
#define NTOT (NCLS * HW)

// A fragment (16x32 bf16): per-lane elements 0..15 -> K = base+0..15
// (base already includes the lane's khalf).
__device__ __forceinline__ bf16x16 frag_a(const float* __restrict__ p, long stride)
{
    bf16x16 f;
#pragma unroll
    for (int j = 0; j < 16; ++j) {
        f[j]     = (__bf16)p[j * stride];
    }
    return f;
}

// B fragment (32x16 bf16): per-lane 16 consecutive K values (base includes kboff).
__device__ __forceinline__ bf16x16 frag_b(const float* __restrict__ p, long stride)
{
    bf16x16 f;
#pragma unroll
    for (int j = 0; j < 16; ++j) f[j] = (__bf16)p[j * stride];
    return f;
}

__device__ __forceinline__ bf16x16 frag_b_mask(const float* __restrict__ p, long stride, float fm)
{
    bf16x16 f;
#pragma unroll
    for (int j = 0; j < 16; ++j) f[j] = (__bf16)(p[j * stride] * fm);
    return f;
}

// ---------------------------------------------------------------------------
// Implicit-GEMM 3x3 'SAME' conv as 9 shifted GEMMs (tap loop fully unrolled,
// boundary handled by clamped address + 0/1 mask -> no EXEC divergence, no
// division in the IC loop). Input (IC x HW) channel-major, output (OC x HW).
// Weights (OC x IC*9) with ALL rows valid (caller zero-pads OC). Optional
// bias / BN / ReLU epilogue; stores guarded to OCr real channels.
// IC must be a multiple of 64.
// ---------------------------------------------------------------------------
__global__ void conv3x3_wmma(const float* __restrict__ W,
                             const float* __restrict__ Xc,
                             const float* __restrict__ bias,
                             const float* __restrict__ bng,
                             const float* __restrict__ bnb,
                             const float* __restrict__ bnm,
                             const float* __restrict__ bnv,
                             float* __restrict__ Out,
                             int OC, int OCr, int IC, int relu)
{
    const int IC9  = IC * 9;
    const int lane = threadIdx.x & 31;
    const int wv   = threadIdx.x >> 5;
    const int mt   = OC >> 4;
    long tile      = (long)blockIdx.x * 8 + wv;
    if (tile >= (long)mt * (HW / 16)) return;

    const int tm = (int)(tile % mt);
    const int tn = (int)(tile / mt);
    const int m0 = tm * 16, n0 = tn * 16;

    const int row   = m0 + (lane & 15);
    const int khalf = (lane >> 4) * 16;
    const int col   = n0 + (lane & 15);
    const int ph    = col / WD, pw = col % WD;
    const int kboff = (lane >> 4) * 16;

    const float* Arow = W + (long)row * IC9 + (long)khalf * 9;

    f32x8 acc = {0.f,0.f,0.f,0.f,0.f,0.f,0.f,0.f};

#pragma unroll
    for (int t = 0; t < 9; ++t) {
        const int rr = t / 3 - 1, cc = t % 3 - 1;   // compile-time after unroll
        int hh = ph + rr, ww = pw + cc;
        float fm = ((unsigned)hh < (unsigned)HD && (unsigned)ww < (unsigned)WD) ? 1.f : 0.f;
        int hc = hh < 0 ? 0 : (hh > HD - 1 ? HD - 1 : hh);
        int wc = ww < 0 ? 0 : (ww > WD - 1 ? WD - 1 : ww);

        const float* Ap = Arow + t;
        const float* Bp = Xc + (long)hc * WD + wc + (long)kboff * HW;

        for (int k0 = 0; k0 < IC; k0 += 64) {
            acc = __builtin_amdgcn_wmma_f32_16x16x32_bf16(
                false, frag_a(Ap, 9), false, frag_b_mask(Bp, HW, fm),
                (short)0, acc, false, false);
            acc = __builtin_amdgcn_wmma_f32_16x16x32_bf16(
                false, frag_a(Ap + 32 * 9, 9), false, frag_b_mask(Bp + 32L * HW, HW, fm),
                (short)0, acc, false, false);
            Ap += 64 * 9;
            Bp += 64L * HW;
        }
    }

    const int n  = n0 + (lane & 15);
    const int mb = m0 + ((lane >> 4) << 3);
#pragma unroll
    for (int r = 0; r < 8; ++r) {
        int m = mb + r;
        if (m < OCr) {
            float val = acc[r];
            if (bias) val += bias[m];
            if (bng) {
                float sc = bng[m] * rsqrtf(bnv[m] + 1e-5f);
                val = (val - bnm[m]) * sc + bnb[m];
            }
            if (relu) val = fmaxf(val, 0.f);
            Out[(long)m * HW + n] = val;
        }
    }
}

// ---------------------------------------------------------------------------
// Generic Out = X @ W^T (+bias)(+ReLU) via WMMA bf16.
//   W: (M x Kd) f32 row-major (all rows valid), X: (Np x Kd) f32 row-major
//   where Np >= ceil16(N) (caller pads) -> guard-free loads.
//   out_nmajor=1 -> Out[n*M + m], else Out[m*N + n]. Kd multiple of 64.
// ---------------------------------------------------------------------------
__global__ void gemm_xwt_wmma(const float* __restrict__ W,
                              const float* __restrict__ X,
                              const float* __restrict__ bias,
                              float* __restrict__ Out,
                              int M, int N, int Kd, int relu, int out_nmajor)
{
    const int lane = threadIdx.x & 31;
    const int wv   = threadIdx.x >> 5;
    const int mt   = M >> 4;
    const int nt   = (N + 15) >> 4;
    long tile      = (long)blockIdx.x * 8 + wv;
    if (tile >= (long)mt * nt) return;

    const int tm = (int)(tile % mt);
    const int tn = (int)(tile / mt);
    const int m0 = tm * 16, n0 = tn * 16;

    const int row   = m0 + (lane & 15);
    const int khalf = (lane >> 4) * 16;
    const int col   = n0 + (lane & 15);
    const int kboff = (lane >> 4) * 16;

    const float* Ap = W + (long)row * Kd + khalf;
    const float* Xp = X + (long)col * Kd + kboff;

    f32x8 acc = {0.f,0.f,0.f,0.f,0.f,0.f,0.f,0.f};
    for (int k0 = 0; k0 < Kd; k0 += 64) {
        acc = __builtin_amdgcn_wmma_f32_16x16x32_bf16(
            false, frag_a(Ap, 1), false, frag_b(Xp, 1), (short)0, acc, false, false);
        acc = __builtin_amdgcn_wmma_f32_16x16x32_bf16(
            false, frag_a(Ap + 32, 1), false, frag_b(Xp + 32, 1), (short)0, acc, false, false);
        Ap += 64;
        Xp += 64;
    }

    const int n  = n0 + (lane & 15);
    const int mb = m0 + ((lane >> 4) << 3);
    if (n < N) {
#pragma unroll
        for (int r = 0; r < 8; ++r) {
            int m = mb + r;
            float val = acc[r];
            if (bias) val += bias[m];
            if (relu) val = fmaxf(val, 0.f);
            if (out_nmajor) Out[(long)n * M + m] = val;
            else            Out[(long)m * N + n] = val;
        }
    }
}

// ---------------------------------------------------------------------------
// Elementwise helpers
// ---------------------------------------------------------------------------
__global__ void pad_rows(const float* __restrict__ src, float* __restrict__ dst,
                         int rows, int cols, int rows_pad)
{
    int i = blockIdx.x * 256 + threadIdx.x;
    if (i >= rows_pad * cols) return;
    dst[i] = (i < rows * cols) ? src[i] : 0.f;
}

__global__ void sigmoid_ip(float* x, int n)
{
    int i = blockIdx.x * 256 + threadIdx.x;
    if (i < n) x[i] = 1.f / (1.f + __expf(-x[i]));
}

__global__ void vadd(float* out, const float* a, const float* b, int n)
{
    int i = blockIdx.x * 256 + threadIdx.x;
    if (i < n) out[i] = a[i] + b[i];
}

// NMS: keep values that equal their 3x3 local max (interior only, classes 8/9
// always kept), everything else -> 0.
__global__ void nms_mask(const float* __restrict__ heat, float* __restrict__ hm)
{
    int i = blockIdx.x * 256 + threadIdx.x;
    if (i >= NTOT) return;
    int c = i / HW, p = i - c * HW;
    float v = heat[i];
    float lm;
    if (c == 8 || c == 9) {
        lm = v;
    } else {
        int y = p / WD, x = p - (p / WD) * WD;
        if (y >= 1 && y <= HD - 2 && x >= 1 && x <= WD - 2) {
            lm = -INFINITY;
#pragma unroll
            for (int dy = -1; dy <= 1; ++dy)
#pragma unroll
                for (int dx = -1; dx <= 1; ++dx)
                    lm = fmaxf(lm, heat[c * HW + (y + dy) * WD + (x + dx)]);
        } else {
            lm = 0.f;
        }
    }
    hm[i] = (v == lm) ? v : 0.f;
}

// ---------------------------------------------------------------------------
// Deterministic top-K=200 over 324000 values (single block, iterative argmax,
// tie-break lowest index — matches jax.lax.top_k ordering). Buffer is L2-hot.
// ---------------------------------------------------------------------------
__global__ void topk_kernel(const float* __restrict__ src,
                            float* __restrict__ scratch,
                            int* __restrict__ top_idx)
{
    const int t = threadIdx.x; // 1024 threads
    for (int i = t; i < NTOT; i += 1024) scratch[i] = src[i];
    __syncthreads();

    __shared__ float sv[1024];
    __shared__ int   si[1024];
    for (int sel = 0; sel < KQ; ++sel) {
        float bv = -INFINITY; int bi = 0x7fffffff;
        for (int i = t; i < NTOT; i += 1024) {
            float v = scratch[i];
            if (v > bv) { bv = v; bi = i; }
        }
        sv[t] = bv; si[t] = bi;
        __syncthreads();
        for (int s = 512; s > 0; s >>= 1) {
            if (t < s) {
                float v2 = sv[t + s]; int i2 = si[t + s];
                if (v2 > sv[t] || (v2 == sv[t] && i2 < si[t])) { sv[t] = v2; si[t] = i2; }
            }
            __syncthreads();
        }
        if (t == 0) { top_idx[sel] = si[0]; scratch[si[0]] = -INFINITY; }
        __syncthreads();
    }
}

// Gather query features + class encoding, derive positions / class / pos idx.
__global__ void build_query(const int* __restrict__ top_idx,
                            const float* __restrict__ lidar,   // (128 x HW)
                            const float* __restrict__ cls_w,   // (128 x 10)
                            const float* __restrict__ cls_b,   // (128)
                            float* __restrict__ x0,            // (200 x 128)
                            float* __restrict__ qpos,          // (200 x 2)
                            int* __restrict__ top_pos,
                            int* __restrict__ top_cls)
{
    int n   = blockIdx.x;          // 200
    int c   = threadIdx.x;         // 128
    int idx = top_idx[n];
    int cls = idx / HW;
    int pos = idx - cls * HW;
    x0[n * HID + c] = lidar[(long)c * HW + pos] + cls_w[c * NCLS + cls] + cls_b[c];
    if (c == 0) {
        int h = pos / WD, w = pos - (pos / WD) * WD;
        qpos[n * 2 + 0] = (float)w + 0.5f;   // reversed (x=j+0.5 first)
        qpos[n * 2 + 1] = (float)h + 0.5f;
        top_pos[n] = pos;
        top_cls[n] = cls;
    }
}

// PE first layer: relu(bn(pos @ w1^T + b1)). synth=1 -> positions derived from
// the flat BEV index n: (w+0.5, h+0.5).
__global__ void pe_layer1(const float* __restrict__ pos,
                          const float* __restrict__ w1, const float* __restrict__ b1,
                          const float* __restrict__ g,  const float* __restrict__ bb,
                          const float* __restrict__ m,  const float* __restrict__ vv,
                          float* __restrict__ out, int N, int synth)
{
    long i = (long)blockIdx.x * 256 + threadIdx.x;
    if (i >= (long)N * HID) return;
    int n = (int)(i >> 7), c = (int)(i & 127);
    float px, py;
    if (synth) {
        int h = n / WD, w = n - (n / WD) * WD;
        px = (float)w + 0.5f; py = (float)h + 0.5f;
    } else {
        px = pos[n * 2]; py = pos[n * 2 + 1];
    }
    float v  = px * w1[c * 2] + py * w1[c * 2 + 1] + b1[c];
    float sc = g[c] * rsqrtf(vv[c] + 1e-5f);
    v = (v - m[c]) * sc + bb[c];
    out[i] = fmaxf(v, 0.f);
}

// kin[n,c] = lidar[c, n] + kpe[n, c]   (cross-attn key/value input)
__global__ void build_kin(const float* __restrict__ lidar,
                          const float* __restrict__ kpe,
                          float* __restrict__ kin)
{
    long i = (long)blockIdx.x * 256 + threadIdx.x;
    if (i >= (long)HW * HID) return;
    int n = (int)(i >> 7), c = (int)(i & 127);
    kin[i] = lidar[(long)c * HW + n] + kpe[i];
}

// ---------------------------------------------------------------------------
// Streaming-softmax multi-head attention: one block per (query, head).
// 256 threads each keep a running (max, sum, acc[16]); LDS tree combine.
// ---------------------------------------------------------------------------
__global__ void mha_attend(const float* __restrict__ QH, int ldq,
                           const float* __restrict__ KH, int ldk,
                           const float* __restrict__ VH, int ldv,
                           float* __restrict__ O, int Nk)
{
    const int q = blockIdx.x >> 3;
    const int h = blockIdx.x & 7;
    const int t = threadIdx.x;         // 256

    float qv[16];
    const float* qp = QH + (long)q * ldq + h * 16;
#pragma unroll
    for (int d = 0; d < 16; ++d) qv[d] = qp[d];

    float mt = -INFINITY, lt = 0.f, acct[16];
#pragma unroll
    for (int d = 0; d < 16; ++d) acct[d] = 0.f;

    for (int j = t; j < Nk; j += 256) {
        const float* kp = KH + (long)j * ldk + h * 16;
        float s = 0.f;
#pragma unroll
        for (int d = 0; d < 16; ++d) s += qv[d] * kp[d];
        s *= 0.25f;                     // 1/sqrt(16)
        float mn = fmaxf(mt, s);
        float fa = __expf(mt - mn);     // mt=-inf -> 0
        float fb = __expf(s - mn);
        lt = lt * fa + fb;
        const float* vp = VH + (long)j * ldv + h * 16;
#pragma unroll
        for (int d = 0; d < 16; ++d) acct[d] = acct[d] * fa + fb * vp[d];
        mt = mn;
    }

    __shared__ float sm[256], sl[256], sa[256][16];
    sm[t] = mt; sl[t] = lt;
#pragma unroll
    for (int d = 0; d < 16; ++d) sa[t][d] = acct[d];
    __syncthreads();
    for (int s = 128; s > 0; s >>= 1) {
        if (t < s) {
            float m1 = sm[t], m2 = sm[t + s];
            float mn = fmaxf(m1, m2);
            float f1 = (sl[t]     > 0.f) ? __expf(m1 - mn) : 0.f;
            float f2 = (sl[t + s] > 0.f) ? __expf(m2 - mn) : 0.f;
            sm[t] = mn;
            sl[t] = sl[t] * f1 + sl[t + s] * f2;
#pragma unroll
            for (int d = 0; d < 16; ++d) sa[t][d] = sa[t][d] * f1 + sa[t + s][d] * f2;
        }
        __syncthreads();
    }
    if (t < 16) O[(long)q * HID + h * 16 + t] = sa[0][t] / sl[0];
}

// LayerNorm over last dim (C=128) of (a+b); one block (128 threads) per row.
__global__ void add_ln(const float* __restrict__ a, const float* __restrict__ b,
                       const float* __restrict__ g, const float* __restrict__ be,
                       float* __restrict__ out)
{
    const int n = blockIdx.x, t = threadIdx.x;   // 128 threads
    __shared__ float r[128];
    float v = a[n * HID + t] + b[n * HID + t];
    r[t] = v; __syncthreads();
    for (int s = 64; s > 0; s >>= 1) { if (t < s) r[t] += r[t + s]; __syncthreads(); }
    float mu = r[0] * (1.f / HID);
    __syncthreads();
    float d = v - mu;
    r[t] = d * d; __syncthreads();
    for (int s = 64; s > 0; s >>= 1) { if (t < s) r[t] += r[t + s]; __syncthreads(); }
    float var = r[0] * (1.f / HID);
    out[n * HID + t] = d * rsqrtf(var + 1e-5f) * g[t] + be[t];
}

// Prediction head: out[c, n] = (relu(bn(x[n] @ w1^T)) @ w2^T + b2)[c]
__global__ void head_kernel(const float* __restrict__ x,     // (200 x 128)
                            const float* __restrict__ w1,    // (64 x 128)
                            const float* __restrict__ g, const float* __restrict__ b,
                            const float* __restrict__ m, const float* __restrict__ v,
                            const float* __restrict__ w2,    // (oc x 64)
                            const float* __restrict__ b2,
                            float* __restrict__ out, int oc) // (oc x 200)
{
    __shared__ float hsh[64];
    const int n = blockIdx.x, t = threadIdx.x;   // 64 threads
    float s = 0.f;
    const float* xr = x + n * HID;
    const float* wr = w1 + t * HID;
    for (int k = 0; k < HID; ++k) s += xr[k] * wr[k];
    float sc = g[t] * rsqrtf(v[t] + 1e-5f);
    s = (s - m[t]) * sc + b[t];
    hsh[t] = fmaxf(s, 0.f);
    __syncthreads();
    if (t < oc) {
        float o = b2[t];
        for (int k = 0; k < 64; ++k) o += hsh[k] * w2[t * 64 + k];
        out[t * KQ + n] = o;
    }
}

__global__ void center_add(float* __restrict__ center, const float* __restrict__ qpos)
{
    int i = blockIdx.x * 256 + threadIdx.x;      // 2*200
    if (i >= 2 * KQ) return;
    int c = i / KQ, n = i - c * KQ;
    center[i] += qpos[n * 2 + c];
}

__global__ void batch_score(const float* __restrict__ hmhead,   // (10 x 200)
                            const float* __restrict__ hmask,    // (10 x HW)
                            const int* __restrict__ top_pos,
                            const int* __restrict__ top_cls,
                            float* __restrict__ out)            // (10 x 200)
{
    int i = blockIdx.x * 256 + threadIdx.x;
    if (i >= NCLS * KQ) return;
    int c = i / KQ, n = i - c * KQ;
    float sg  = 1.f / (1.f + __expf(-hmhead[i]));
    float qhs = hmask[(long)c * HW + top_pos[n]];
    float oh  = (top_cls[n] == c) ? 1.f : 0.f;
    out[i] = sg * qhs * oh;
}

// ---------------------------------------------------------------------------
// Input indices per setup_inputs() dict order (recursive insertion order)
// ---------------------------------------------------------------------------
enum {
    IN_FEAT = 0, IN_SHW, IN_SHB, IN_HMW,
    IN_HMG, IN_HMB, IN_HMM, IN_HMV,
    IN_HOW, IN_HOB, IN_CLSW, IN_CLSB,
    IN_SPE_W1, IN_SPE_B1, IN_SPE_G, IN_SPE_BB, IN_SPE_M, IN_SPE_V, IN_SPE_W2, IN_SPE_B2,
    IN_CPE_W1, IN_CPE_B1, IN_CPE_G, IN_CPE_BB, IN_CPE_M, IN_CPE_V, IN_CPE_W2, IN_CPE_B2,
    IN_SA_WIN, IN_SA_BIN, IN_SA_WOUT, IN_SA_BOUT,
    IN_CA_WIN, IN_CA_BIN, IN_CA_WOUT, IN_CA_BOUT,
    IN_L1W, IN_L1B, IN_L2W, IN_L2B,
    IN_N1G, IN_N1B, IN_N2G, IN_N2B, IN_N3G, IN_N3B,
    IN_HEADS // 46; per head: w1, bn.g, bn.b, bn.m, bn.v, w2, b2 (7 each)
};

extern "C" void kernel_launch(void* const* d_in, const int* in_sizes, int n_in,
                              void* d_out, int out_size, void* d_ws, size_t ws_size,
                              hipStream_t stream)
{
    (void)in_sizes; (void)n_in; (void)out_size; (void)ws_size;
    const float* F = (const float*)d_in[IN_FEAT];
    auto in = [&](int i) { return (const float*)d_in[i]; };

    // ---- workspace bump allocator (reused buffers annotated) ----
    size_t off = 0;
    auto alloc = [&](size_t bytes) -> void* {
        void* p = (char*)d_ws + off;
        off += (bytes + 255) & ~(size_t)255;
        return p;
    };
    const size_t BIG = (size_t)HID * HW * sizeof(float); // 16.6 MB
    float* lidar = (float*)alloc(BIG);                   // shared conv output
    float* bufA  = (float*)alloc(BIG);                   // hmid  -> kpe
    float* bufB  = (float*)alloc(BIG);                   // h1c   -> kh
    float* bufC  = (float*)alloc(BIG);                   // vh
    float* bufD  = (float*)alloc(BIG);                   // kin
    float* heat  = (float*)alloc((size_t)NTOT * 4);      // dense hm -> sigmoid (in place)
    float* hmask = (float*)alloc((size_t)NTOT * 4);
    float* tscr  = (float*)alloc((size_t)NTOT * 4);
    float* wpad  = (float*)alloc((size_t)16 * 1152 * 4); // zero-padded hm_out weights
    int*   top_idx = (int*)alloc(256 * 4);
    int*   top_pos = (int*)alloc(256 * 4);
    int*   top_cls = (int*)alloc(256 * 4);
    float* qpos  = (float*)alloc(KQ * 2 * 4);
    float* x0    = (float*)alloc(KQP * HID * 4);
    float* qpe   = (float*)alloc(KQP * HID * 4);
    float* h1s   = (float*)alloc(KQP * HID * 4);
    float* qin1  = (float*)alloc(KQP * HID * 4);
    float* qkv   = (float*)alloc(KQP * 384 * 4);
    float* o_s   = (float*)alloc(KQP * HID * 4);
    float* a1    = (float*)alloc(KQP * HID * 4);
    float* x1    = (float*)alloc(KQP * HID * 4);
    float* qin2  = (float*)alloc(KQP * HID * 4);
    float* qh_c  = (float*)alloc(KQP * HID * 4);
    float* o_c   = (float*)alloc(KQP * HID * 4);
    float* a2    = (float*)alloc(KQP * HID * 4);
    float* x2    = (float*)alloc(KQP * HID * 4);
    float* f1    = (float*)alloc(KQP * 256 * 4);
    float* f2    = (float*)alloc(KQP * HID * 4);
    float* x3    = (float*)alloc(KQP * HID * 4);
    float* hmh   = (float*)alloc(NCLS * KQ * 4);
    float* out   = (float*)d_out;

    auto gemm = [&](const float* W, const float* X, const float* bias, float* O,
                    int M, int N, int Kd, int relu, int nmajor) {
        int ntiles = (M / 16) * ((N + 15) / 16);
        gemm_xwt_wmma<<<(ntiles + 7) / 8, 256, 0, stream>>>(W, X, bias, O,
                                                            M, N, Kd, relu, nmajor);
    };

    // ---- 0. zero-pad hm_out weights 10x1152 -> 16x1152 ----
    pad_rows<<<(16 * 1152 + 255) / 256, 256, 0, stream>>>(in(IN_HOW), wpad, 10, 1152, 16);
    // ---- 1. shared 3x3 conv 512->128 (+bias) ----
    conv3x3_wmma<<<(8 * (HW / 16) + 7) / 8, 256, 0, stream>>>(
        in(IN_SHW), F, in(IN_SHB), nullptr, nullptr, nullptr, nullptr,
        lidar, 128, 128, 512, 0);
    // ---- 2. hm block conv 128->128 + BN + ReLU ----
    conv3x3_wmma<<<(8 * (HW / 16) + 7) / 8, 256, 0, stream>>>(
        in(IN_HMW), lidar, nullptr, in(IN_HMG), in(IN_HMB), in(IN_HMM), in(IN_HMV),
        bufA, 128, 128, 128, 1);
    // ---- 3. hm out conv 128->10 (+bias), OC padded to 16 ----
    conv3x3_wmma<<<((HW / 16) + 7) / 8, 256, 0, stream>>>(
        wpad, bufA, in(IN_HOB), nullptr, nullptr, nullptr, nullptr,
        heat, 16, NCLS, 128, 0);
    // ---- 4. sigmoid + NMS mask ----
    sigmoid_ip<<<(NTOT + 255) / 256, 256, 0, stream>>>(heat, NTOT);
    nms_mask<<<(NTOT + 255) / 256, 256, 0, stream>>>(heat, hmask);
    // ---- 5. top-K=200 ----
    topk_kernel<<<1, 1024, 0, stream>>>(hmask, tscr, top_idx);
    // ---- 6. gather queries + class encoding ----
    build_query<<<KQ, HID, 0, stream>>>(top_idx, lidar, in(IN_CLSW), in(IN_CLSB),
                                        x0, qpos, top_pos, top_cls);
    // ---- 7. positional encodings ----
    pe_layer1<<<(KQ * HID + 255) / 256, 256, 0, stream>>>(
        qpos, in(IN_SPE_W1), in(IN_SPE_B1), in(IN_SPE_G), in(IN_SPE_BB),
        in(IN_SPE_M), in(IN_SPE_V), h1s, KQ, 0);
    gemm(in(IN_SPE_W2), h1s, in(IN_SPE_B2), qpe, HID, KQ, HID, 0, 1);
    pe_layer1<<<((long)HW * HID + 255) / 256, 256, 0, stream>>>(
        nullptr, in(IN_CPE_W1), in(IN_CPE_B1), in(IN_CPE_G), in(IN_CPE_BB),
        in(IN_CPE_M), in(IN_CPE_V), bufB, HW, 1);
    gemm(in(IN_CPE_W2), bufB, in(IN_CPE_B2), bufA /*kpe*/, HID, HW, HID, 0, 1);
    build_kin<<<((long)HW * HID + 255) / 256, 256, 0, stream>>>(lidar, bufA, bufD);
    // ---- 8. self-attention ----
    vadd<<<(KQ * HID + 255) / 256, 256, 0, stream>>>(qin1, x0, qpe, KQ * HID);
    gemm(in(IN_SA_WIN), qin1, in(IN_SA_BIN), qkv, 384, KQ, HID, 0, 1);
    mha_attend<<<KQ * 8, 256, 0, stream>>>(qkv, 384, qkv + 128, 384, qkv + 256, 384,
                                           o_s, KQ);
    gemm(in(IN_SA_WOUT), o_s, in(IN_SA_BOUT), a1, HID, KQ, HID, 0, 1);
    add_ln<<<KQ, HID, 0, stream>>>(x0, a1, in(IN_N1G), in(IN_N1B), x1);
    // ---- 9. cross-attention ----
    vadd<<<(KQ * HID + 255) / 256, 256, 0, stream>>>(qin2, x1, qpe, KQ * HID);
    gemm(in(IN_CA_WIN),                 qin2, in(IN_CA_BIN),       qh_c, HID, KQ, HID, 0, 1);
    gemm(in(IN_CA_WIN) + 128 * HID, bufD, in(IN_CA_BIN) + 128, bufB /*kh*/, HID, HW, HID, 0, 1);
    gemm(in(IN_CA_WIN) + 256 * HID, bufD, in(IN_CA_BIN) + 256, bufC /*vh*/, HID, HW, HID, 0, 1);
    mha_attend<<<KQ * 8, 256, 0, stream>>>(qh_c, HID, bufB, HID, bufC, HID, o_c, HW);
    gemm(in(IN_CA_WOUT), o_c, in(IN_CA_BOUT), a2, HID, KQ, HID, 0, 1);
    add_ln<<<KQ, HID, 0, stream>>>(x1, a2, in(IN_N2G), in(IN_N2B), x2);
    // ---- 10. FFN ----
    gemm(in(IN_L1W), x2, in(IN_L1B), f1, 256, KQ, HID, 1, 1);
    gemm(in(IN_L2W), f1, in(IN_L2B), f2, HID, KQ, 256, 0, 1);
    add_ln<<<KQ, HID, 0, stream>>>(x2, f2, in(IN_N3G), in(IN_N3B), x3);
    // ---- 11. heads ----
    // d_out layout: score[0:2000] rot[2000:2400] dim[2400:3000]
    //               center[3000:3400] height[3600:3800] vel[3800:4200]
    struct { int hidx; int oc; float* dst; } heads[6] = {
        {0, 2,  out + 3000},  // center
        {1, 1,  out + 3600},  // height
        {2, 3,  out + 2400},  // dim
        {3, 2,  out + 2000},  // rot
        {4, 2,  out + 3800},  // vel
        {5, 10, hmh},         // heatmap
    };
    for (int hh = 0; hh < 6; ++hh) {
        int b = IN_HEADS + heads[hh].hidx * 7;
        head_kernel<<<KQ, 64, 0, stream>>>(x3, in(b + 0), in(b + 1), in(b + 2),
                                           in(b + 3), in(b + 4), in(b + 5), in(b + 6),
                                           heads[hh].dst, heads[hh].oc);
    }
    center_add<<<(2 * KQ + 255) / 256, 256, 0, stream>>>(out + 3000, qpos);
    batch_score<<<(NCLS * KQ + 255) / 256, 256, 0, stream>>>(hmh, hmask, top_pos,
                                                             top_cls, out);
}
